// DySample_86045374808258
// MI455X (gfx1250) — matrix operand order repaired and checked by reading, not compiled, not run on hardware
//
#include <hip/hip_runtime.h>
#include <hip/hip_bf16.h>
#include <math.h>

// DySample fused: (1) offset/scope GEMM -> coords, (2) bilinear sample + end GEMM.
// All matrix math via v_wmma_f32_16x16x32_bf16 (CDNA5 / gfx1250, wave32).

typedef __attribute__((ext_vector_type(16))) __bf16 v16bf;
typedef __attribute__((ext_vector_type(8)))  float  v8f;

#define BATCH 8
#define CIN   128
#define W_IN  128
#define H_IN  128
#define HW    16384      // 128*128
#define SHW   65536      // 256*256

// Dense 16-bit A-matrix (16x32) K index for element e of the v16bf fragment.
// Contiguous order matching B: K = e, upper half-wave (lanes 16-31) adds +16.
__device__ __forceinline__ int a_kmap(int e, int half) {
    return e + half * 16;
}

// ---------------------------------------------------------------- kernel 0
// Convert weights to bf16 once: w1 = [offset_w ; scope_w] (64x128), w2 = end_w (128x128).
__global__ __launch_bounds__(256) void dys_convert_weights(
    const float* __restrict__ offset_w, const float* __restrict__ scope_w,
    const float* __restrict__ end_w, __bf16* __restrict__ w1, __bf16* __restrict__ w2)
{
    int gid = blockIdx.x * 256 + threadIdx.x;   // grid covers 8192 + 16384 = 24576
    if (gid < 8192) {
        int row = gid >> 7, k = gid & 127;
        float v = (row < 32) ? offset_w[row * 128 + k] : scope_w[(row - 32) * 128 + k];
        w1[gid] = (__bf16)v;
    } else {
        int i = gid - 8192;
        w2[i] = (__bf16)end_w[i];
    }
}

// ---------------------------------------------------------------- kernel 1
// Per (b, 64-pixel tile): D(64x64) = W1(64x128) * X(128x64) via WMMA bf16,
// then coords[b, q, hw] = (w + offx*sig(scx)*0.5 + initx, h + offy*sig(scy)*0.5 + inity).
__global__ __launch_bounds__(256) void dys_offsets(
    const float* __restrict__ x, const float* __restrict__ offset_b,
    const __bf16* __restrict__ w1, float2* __restrict__ coords)
{
    __shared__ __align__(32) __bf16 xT[64 * 144];   // [n][k], padded row = 144 (288B, 32B-aligned)
    __shared__ float dstage[64 * 68];               // [M][n] staging for cross-wave combine

    const int t   = threadIdx.x;
    const int b   = blockIdx.y;
    const int hw0 = blockIdx.x * 64;

    // Load X tile transposed: xT[n][k] = bf16(x[b, k, hw0+n]); coalesced global reads.
    const float* xb = x + (size_t)b * CIN * HW + hw0;
    for (int it = 0; it < 32; ++it) {
        int lin = it * 256 + t;
        int k = lin >> 6;
        int n = lin & 63;
        xT[n * 144 + k] = (__bf16)xb[(size_t)k * HW + n];
    }
    __syncthreads();

    const int lane = t & 31;
    const int wv   = t >> 5;            // 8 waves
    const int m_tile = wv & 3;          // M tiles 0..3 (M=64)
    const int half = lane >> 4;
    const int l15  = lane & 15;

    // A fragments: rows of w1 (L2-resident bf16), contiguous k order (k = e + 16*half) matching B.
    v16bf a[4];
    for (int kk = 0; kk < 4; ++kk) {
        v16bf av;
        const __bf16* wrow = w1 + (m_tile * 16 + l15) * 128 + kk * 32;
        for (int e = 0; e < 16; ++e) av[e] = wrow[a_kmap(e, half)];
        a[kk] = av;
    }

    for (int s = 0; s < 2; ++s) {
        int nsub = (wv >> 2) * 2 + s;   // waves 0-3 -> nsub 0,1 ; waves 4-7 -> 2,3
        v8f acc = {};
        for (int kk = 0; kk < 4; ++kk) {
            v16bf bf = *(const v16bf*)&xT[(nsub * 16 + l15) * 144 + kk * 32 + half * 16];
            acc = __builtin_amdgcn_wmma_f32_16x16x32_bf16(false, a[kk], false, bf,
                                                          (short)0, acc, false, false);
        }
        for (int r = 0; r < 8; ++r)
            dstage[(m_tile * 16 + r + half * 8) * 68 + nsub * 16 + l15] = acc[r];
    }
    __syncthreads();

    // coords: rows 0-15 off_x, 16-31 off_y, 32-47 scope_x, 48-63 scope_y.
    for (int it = 0; it < 4; ++it) {
        int pid = it * 256 + t;         // 16 q * 64 n = 1024
        int q = pid >> 6;
        int n = pid & 63;
        float offx = dstage[q * 68 + n] + offset_b[q];
        float offy = dstage[(16 + q) * 68 + n] + offset_b[16 + q];
        float sgx = 1.0f / (1.0f + __expf(-dstage[(32 + q) * 68 + n]));
        float sgy = 1.0f / (1.0f + __expf(-dstage[(48 + q) * 68 + n]));
        int p = q & 3;
        int ii = p >> 1, jj = p & 1;    // sub-pixel (i,j); init_x = h[j], init_y = h[i]
        int hw = hw0 + n;
        int wpix = hw & (W_IN - 1);
        int hpix = hw >> 7;
        float sx = (float)wpix + offx * sgx * 0.5f + (jj ? 0.25f : -0.25f);
        float sy = (float)hpix + offy * sgy * 0.5f + (ii ? 0.25f : -0.25f);
        coords[(size_t)(b * 16 + q) * HW + hw] = make_float2(sx, sy);
    }
}

// ---------------------------------------------------------------- kernel 2
// Per (b, H2, 64-wide W2 tile): gather bilinear samples (x is L2-resident) into a
// bf16 LDS tile [n][c], then out = end_w * V + end_b via WMMA bf16.
__global__ __launch_bounds__(256) void dys_sample_gemm(
    const float* __restrict__ x, const float2* __restrict__ coords,
    const __bf16* __restrict__ w2, const float* __restrict__ end_b,
    float* __restrict__ out)
{
    __shared__ __align__(32) __bf16 smp[64 * 144];  // [n][c], padded row (288B)

    const int t    = threadIdx.x;
    const int b    = blockIdx.z;
    const int H2   = blockIdx.y;
    const int W2_0 = blockIdx.x * 64;

    // Gather: thread t -> pixel n = t>>2, group gi = t&3; 32 channels each.
    {
        int n  = t >> 2;
        int gi = t & 3;
        int W2 = W2_0 + n;
        int h = H2 >> 1, ii = H2 & 1;
        int wq = W2 >> 1, jj = W2 & 1;
        int q = gi * 4 + ii * 2 + jj;
        float2 crd = coords[(size_t)(b * 16 + q) * HW + h * W_IN + wq];
        float fx0 = floorf(crd.x), fy0 = floorf(crd.y);
        float fx = crd.x - fx0, fy = crd.y - fy0;
        int ix0 = (int)fx0, iy0 = (int)fy0;
        int ix1 = ix0 + 1, iy1 = iy0 + 1;
        float w00 = (1.f - fy) * (1.f - fx), w01 = (1.f - fy) * fx;
        float w10 = fy * (1.f - fx),         w11 = fy * fx;
        bool vx0 = (ix0 >= 0) & (ix0 < W_IN), vx1 = (ix1 >= 0) & (ix1 < W_IN);
        bool vy0 = (iy0 >= 0) & (iy0 < H_IN), vy1 = (iy1 >= 0) & (iy1 < H_IN);
        w00 = (vx0 && vy0) ? w00 : 0.f;
        w01 = (vx1 && vy0) ? w01 : 0.f;
        w10 = (vx0 && vy1) ? w10 : 0.f;
        w11 = (vx1 && vy1) ? w11 : 0.f;
        int cx0 = min(max(ix0, 0), W_IN - 1), cx1 = min(max(ix1, 0), W_IN - 1);
        int cy0 = min(max(iy0, 0), H_IN - 1), cy1 = min(max(iy1, 0), H_IN - 1);
        int o00 = cy0 * W_IN + cx0, o01 = cy0 * W_IN + cx1;
        int o10 = cy1 * W_IN + cx0, o11 = cy1 * W_IN + cx1;
        const float* base = x + (size_t)(b * CIN + gi * 32) * HW;
        __bf16* srow = &smp[n * 144 + gi * 32];
        for (int c = 0; c < 32; ++c) {
            const float* p = base + (size_t)c * HW;
            float v = w00 * p[o00] + w01 * p[o01] + w10 * p[o10] + w11 * p[o11];
            srow[c] = (__bf16)v;
        }
    }
    __syncthreads();

    const int lane = t & 31;
    const int wv   = t >> 5;      // wave = M tile (0..7), M = 128
    const int half = lane >> 4;
    const int l15  = lane & 15;

    v16bf a[4];
    for (int kk = 0; kk < 4; ++kk) {
        v16bf av;
        const __bf16* wrow = w2 + (wv * 16 + l15) * 128 + kk * 32;
        for (int e = 0; e < 16; ++e) av[e] = wrow[a_kmap(e, half)];
        a[kk] = av;
    }
    float bias[8];
    for (int r = 0; r < 8; ++r) bias[r] = end_b[wv * 16 + r + half * 8];

    float* outb = out + (size_t)b * CIN * SHW + (size_t)H2 * 256 + W2_0;
    for (int nsub = 0; nsub < 4; ++nsub) {
        v8f acc = {};
        for (int kk = 0; kk < 4; ++kk) {
            v16bf bf = *(const v16bf*)&smp[(nsub * 16 + l15) * 144 + kk * 32 + half * 16];
            acc = __builtin_amdgcn_wmma_f32_16x16x32_bf16(false, a[kk], false, bf,
                                                          (short)0, acc, false, false);
        }
        for (int r = 0; r < 8; ++r) {
            int M = wv * 16 + r + half * 8;
            outb[(size_t)M * SHW + nsub * 16 + l15] = acc[r] + bias[r];
        }
    }
}

// ---------------------------------------------------------------- launch
extern "C" void kernel_launch(void* const* d_in, const int* in_sizes, int n_in,
                              void* d_out, int out_size, void* d_ws, size_t ws_size,
                              hipStream_t stream) {
    const float* x        = (const float*)d_in[0];
    const float* offset_w = (const float*)d_in[1];
    const float* offset_b = (const float*)d_in[2];
    const float* scope_w  = (const float*)d_in[3];
    const float* end_w    = (const float*)d_in[4];
    const float* end_b    = (const float*)d_in[5];
    float*       out      = (float*)d_out;

    char* ws = (char*)d_ws;
    __bf16* w1 = (__bf16*)ws;                       // 64*128 bf16 = 16 KB
    __bf16* w2 = (__bf16*)(ws + 16384);             // 128*128 bf16 = 32 KB
    float2* coords = (float2*)(ws + 65536);         // 8*16*16384 float2 = 16 MB

    dys_convert_weights<<<96, 256, 0, stream>>>(offset_w, scope_w, end_w, w1, w2);
    dys_offsets<<<dim3(HW / 64, BATCH), 256, 0, stream>>>(x, offset_b, w1, coords);
    dys_sample_gemm<<<dim3(4, 256, BATCH), 256, 0, stream>>>(x, coords, w2, end_b, out);
}
